// GCN_79740362817955
// MI455X (gfx1250) — matrix operand order repaired and checked by reading, not compiled, not run on hardware
//
#include <hip/hip_runtime.h>

typedef __attribute__((ext_vector_type(16))) __bf16 v16bf;
typedef __attribute__((ext_vector_type(8)))  float  v8f;

#define N_NODES     50000
#define NUM_FEAT    5
#define HIDDEN      256
#define NUM_CLASSES 5
#define N_EDGES     800000
#define KBIG        512   // [agg2 | h1] concatenated K dimension

// ---------- bf16 helpers (RNE) ----------
static __device__ __forceinline__ unsigned short f2bf(float f) {
    unsigned int u = __float_as_uint(f);
    u += 0x7FFFu + ((u >> 16) & 1u);
    return (unsigned short)(u >> 16);
}
static __device__ __forceinline__ float bf2f(unsigned short h) {
    return __uint_as_float(((unsigned int)h) << 16);
}

// ---------- K1: layer-1 edge scatter (msg sum + degree) ----------
__global__ __launch_bounds__(256) void sage_edge_l1(
    const long long* __restrict__ ei, const float* __restrict__ x,
    float* __restrict__ agg1, float* __restrict__ deg)
{
    int e = blockIdx.x * 256 + threadIdx.x;
    if (e >= N_EDGES) return;
    int s = (int)ei[e];
    int d = (int)ei[N_EDGES + e];
    const float* xs = x + (size_t)s * NUM_FEAT;
    float* ad = agg1 + (size_t)d * NUM_FEAT;
#pragma unroll
    for (int f = 0; f < NUM_FEAT; ++f) atomicAdd(&ad[f], xs[f]);
    atomicAdd(&deg[d], 1.0f);
}

// ---------- K2: layer-1 node update: h1 = relu(agg@W1_l + b1 + x@W1_r) -> bf16 ----------
__global__ __launch_bounds__(256) void sage_node_l1(
    const float* __restrict__ x, const float* __restrict__ agg1,
    const float* __restrict__ deg,
    const float* __restrict__ W1l, const float* __restrict__ b1,
    const float* __restrict__ W1r, unsigned short* __restrict__ Abig)
{
    __shared__ float xa[NUM_FEAT], aa[NUM_FEAT];
    int n = blockIdx.x, c = threadIdx.x;
    if (c < NUM_FEAT) {
        xa[c] = x[(size_t)n * NUM_FEAT + c];
        float dg = deg[n]; dg = dg > 1.f ? dg : 1.f;
        aa[c] = agg1[(size_t)n * NUM_FEAT + c] / dg;
    }
    __syncthreads();
    float acc = b1[c];
#pragma unroll
    for (int k = 0; k < NUM_FEAT; ++k)
        acc += aa[k] * W1l[k * HIDDEN + c] + xa[k] * W1r[k * HIDDEN + c];
    acc = acc > 0.f ? acc : 0.f;
    // h1 lives in cols [256,512) of the concatenated bf16 A matrix
    Abig[(size_t)n * KBIG + HIDDEN + c] = f2bf(acc);
}

// ---------- K3: pack W2 into bf16 WT[256 out][512 in] = [W2_l ; W2_r]^T ----------
__global__ __launch_bounds__(256) void sage_prep_w2(
    const float* __restrict__ W2l, const float* __restrict__ W2r,
    unsigned short* __restrict__ WT)
{
    int idx = blockIdx.x * 256 + threadIdx.x;     // 0 .. 256*512-1
    int nout = idx >> 9;
    int k    = idx & (KBIG - 1);
    float w = (k < HIDDEN) ? W2l[(size_t)k * HIDDEN + nout]
                           : W2r[(size_t)(k - HIDDEN) * HIDDEN + nout];
    WT[(size_t)nout * KBIG + k] = f2bf(w);
}

// ---------- K4: layer-2 edge scatter: agg2[dst] += h1[src] (bf16 gather, f32 atomics) ----------
__global__ __launch_bounds__(256) void sage_edge_l2(
    const long long* __restrict__ ei, const unsigned short* __restrict__ Abig,
    float* __restrict__ agg2)
{
    int e    = (blockIdx.x * 256 + threadIdx.x) >> 5;   // one edge per wave32
    int lane = threadIdx.x & 31;
    if (e >= N_EDGES) return;
    int s = (int)ei[e];
    int d = (int)ei[N_EDGES + e];
    const unsigned short* hrow = Abig + (size_t)s * KBIG + HIDDEN;
    float* arow = agg2 + (size_t)d * HIDDEN;
#pragma unroll
    for (int i = 0; i < 8; ++i) {
        int f = lane + i * 32;
        atomicAdd(&arow[f], bf2f(hrow[f]));
    }
}

// ---------- K5: normalize agg2 by degree, convert to bf16 into A cols [0,256) ----------
__global__ __launch_bounds__(256) void sage_norm_l2(
    const float* __restrict__ agg2, const float* __restrict__ deg,
    unsigned short* __restrict__ Abig)
{
    size_t idx = (size_t)blockIdx.x * 256 + threadIdx.x;
    int n = (int)(idx >> 8);
    int f = (int)(idx & 255);
    float dg = deg[n]; dg = dg > 1.f ? dg : 1.f;
    Abig[(size_t)n * KBIG + f] = f2bf(agg2[idx] / dg);
}

// ---------- K6: WMMA GEMM: h2 = relu(Abig[50000,512](bf16) @ W[512,256] + b2) ----------
// One 16x16 output tile per wave; K unrolled in steps of 32 (v_wmma_f32_16x16x32_bf16).
// A fragment (16-bit A 16x32 layout): lane L (row = L&15, half = L>>4) holds
//   elems 0-7  -> K = k0 + 8*half + e
//   elems 8-15 -> K = k0 + 16 + 8*half + (e-8)
// B fragment (K x N, N = lane&15): same K mapping as A  (WT row-contig)
__global__ __launch_bounds__(256) void sage_gemm_wmma(
    const unsigned short* __restrict__ Abig,  // [N_NODES][512]
    const unsigned short* __restrict__ WT,    // [256][512]
    const float* __restrict__ b2,
    float* __restrict__ h2)                   // [N_NODES][256]
{
    union Frag { v16bf v; uint4 q[2]; };
    const int lane = threadIdx.x & 31;
    const int wave = threadIdx.x >> 5;
    const int m0   = blockIdx.x * 16;
    const int n0   = blockIdx.y * 128 + wave * 16;
    const int half = lane >> 4;
    const int l15  = lane & 15;

    const unsigned short* Arow = Abig + (size_t)(m0 + l15) * KBIG;
    const unsigned short* Brow = WT   + (size_t)(n0 + l15) * KBIG;

    v8f acc = {};
#pragma unroll
    for (int k0 = 0; k0 < KBIG; k0 += 32) {
        Frag a, b;
        const int ka = k0 + 8 * half;
        a.q[0] = *(const uint4*)(Arow + ka);        // 8 bf16, 16B aligned
        a.q[1] = *(const uint4*)(Arow + ka + 16);
        const int kb = k0 + 8 * half;
        b.q[0] = *(const uint4*)(Brow + kb);
        b.q[1] = *(const uint4*)(Brow + kb + 16);
        acc = __builtin_amdgcn_wmma_f32_16x16x32_bf16(
            /*neg_a=*/false, a.v, /*neg_b=*/false, b.v,
            /*c_mod=*/(short)0, acc, /*reuse_a=*/false, /*reuse_b=*/false);
    }

    // C/D layout: col = lane&15; VGPR r -> row m0 + r + 8*half
    const int col   = n0 + l15;
    const float bias = b2[col];
    const int rbase = m0 + 8 * half;
#pragma unroll
    for (int r = 0; r < 8; ++r) {
        float v = acc[r] + bias;
        h2[(size_t)(rbase + r) * HIDDEN + col] = v > 0.f ? v : 0.f;
    }
}

// ---------- K7: out = softmax(h2 @ W_out + b_out), one wave32 per node ----------
__global__ __launch_bounds__(256) void sage_out_softmax(
    const float* __restrict__ h2, const float* __restrict__ Wout,
    const float* __restrict__ bout, float* __restrict__ out)
{
    const int lane = threadIdx.x & 31;
    const int node = blockIdx.x * 8 + (threadIdx.x >> 5);
    if (node >= N_NODES) return;
    const float* hrow = h2 + (size_t)node * HIDDEN;
    float p0 = 0, p1 = 0, p2 = 0, p3 = 0, p4 = 0;
    for (int k = lane; k < HIDDEN; k += 32) {
        float h = hrow[k];
        const float* w = Wout + (size_t)k * NUM_CLASSES;
        p0 += h * w[0]; p1 += h * w[1]; p2 += h * w[2];
        p3 += h * w[3]; p4 += h * w[4];
    }
#pragma unroll
    for (int off = 16; off; off >>= 1) {
        p0 += __shfl_xor(p0, off, 32);
        p1 += __shfl_xor(p1, off, 32);
        p2 += __shfl_xor(p2, off, 32);
        p3 += __shfl_xor(p3, off, 32);
        p4 += __shfl_xor(p4, off, 32);
    }
    if (lane == 0) {
        float l[NUM_CLASSES] = { p0 + bout[0], p1 + bout[1], p2 + bout[2],
                                 p3 + bout[3], p4 + bout[4] };
        float m = l[0];
#pragma unroll
        for (int c = 1; c < NUM_CLASSES; ++c) m = l[c] > m ? l[c] : m;
        float s = 0.f;
#pragma unroll
        for (int c = 0; c < NUM_CLASSES; ++c) { l[c] = __expf(l[c] - m); s += l[c]; }
        float inv = 1.f / s;
#pragma unroll
        for (int c = 0; c < NUM_CLASSES; ++c)
            out[(size_t)node * NUM_CLASSES + c] = l[c] * inv;
    }
}

extern "C" void kernel_launch(void* const* d_in, const int* in_sizes, int n_in,
                              void* d_out, int out_size, void* d_ws, size_t ws_size,
                              hipStream_t stream)
{
    (void)in_sizes; (void)n_in; (void)out_size; (void)ws_size;
    const float*     x    = (const float*)d_in[0];
    const long long* ei   = (const long long*)d_in[1];   // int64 [2, N_EDGES]
    const float*     W1l  = (const float*)d_in[2];
    const float*     b1   = (const float*)d_in[3];
    const float*     W1r  = (const float*)d_in[4];
    const float*     W2l  = (const float*)d_in[5];
    const float*     b2   = (const float*)d_in[6];
    const float*     W2r  = (const float*)d_in[7];
    const float*     Wout = (const float*)d_in[8];
    const float*     bout = (const float*)d_in[9];
    float* out = (float*)d_out;

    // workspace carve-up (256B aligned)
    char* ws = (char*)d_ws;
    size_t off = 0;
    auto carve = [&](size_t bytes) -> char* {
        char* p = ws + off;
        off += (bytes + 255) & ~(size_t)255;
        return p;
    };
    float* deg            = (float*)carve((size_t)N_NODES * 4);
    float* agg1           = (float*)carve((size_t)N_NODES * NUM_FEAT * 4);
    float* agg2           = (float*)carve((size_t)N_NODES * HIDDEN * 4);  // reused as h2
    size_t zero_bytes     = off;                                          // atomic targets
    unsigned short* Abig  = (unsigned short*)carve((size_t)N_NODES * KBIG * 2);
    unsigned short* WT    = (unsigned short*)carve((size_t)HIDDEN * KBIG * 2);
    float* h2 = agg2;  // safe alias: agg2 fully consumed into Abig before GEMM writes h2

    hipMemsetAsync(d_ws, 0, zero_bytes, stream);

    sage_edge_l1<<<(N_EDGES + 255) / 256, 256, 0, stream>>>(ei, x, agg1, deg);
    sage_node_l1<<<N_NODES, HIDDEN, 0, stream>>>(x, agg1, deg, W1l, b1, W1r, Abig);
    sage_prep_w2<<<(HIDDEN * KBIG) / 256, 256, 0, stream>>>(W2l, W2r, WT);
    sage_edge_l2<<<(N_EDGES * 32) / 256, 256, 0, stream>>>(ei, Abig, agg2);
    sage_norm_l2<<<((size_t)N_NODES * HIDDEN) / 256, 256, 0, stream>>>(agg2, deg, Abig);
    sage_gemm_wmma<<<dim3(N_NODES / 16, 2), 256, 0, stream>>>(Abig, WT, b2, h2);
    sage_out_softmax<<<(N_NODES + 7) / 8, 256, 0, stream>>>(h2, Wout, bout, out);
}
